// GPT_10445360464227
// MI455X (gfx1250) — matrix operand order repaired and checked by reading, not compiled, not run on hardware
//
#include <hip/hip_runtime.h>
#include <math.h>

// ---------------- model constants (match reference) ----------------
#define Vv 32000
#define Ll 8
#define Hh 16
#define Cc 1024
#define Bb 4
#define Tt 1024
#define Dd 64          // head dim
#define MT (Bb*Tt)     // 4096 rows of activations
#define FOURC (4*Cc)

// ---------------- types / helpers ----------------
typedef __attribute__((ext_vector_type(16))) __bf16 v16bf;
typedef __attribute__((ext_vector_type(8)))  float  v8f;
typedef __attribute__((ext_vector_type(2)))  float  v2f;
typedef __attribute__((ext_vector_type(2)))  __bf16 v2bf;
typedef __attribute__((ext_vector_type(4)))  int    v4i;

union BF16x16 { v16bf v; unsigned short u[16]; };

// single v_cvt_pk_bf16_f32
__device__ inline unsigned int cvt2bf(float a, float b) {
    v2f f; f.x = a; f.y = b;
    v2bf h = __builtin_convertvector(f, v2bf);
    return __builtin_bit_cast(unsigned int, h);
}
__device__ inline unsigned short f2bfh(float a) {
    return __builtin_bit_cast(unsigned short, (__bf16)a);
}

__device__ inline float wred_sum(float v) {
    #pragma unroll
    for (int o = 16; o > 0; o >>= 1) v += __shfl_xor(v, o, 32);
    return v;
}
__device__ inline float wred_max(float v) {
    #pragma unroll
    for (int o = 16; o > 0; o >>= 1) v = fmaxf(v, __shfl_xor(v, o, 32));
    return v;
}

// ---------------- async global->LDS (CDNA5), with portable fallback ----------------
#if defined(__has_builtin)
# if __has_builtin(__builtin_amdgcn_global_load_async_to_lds_b128) && __has_builtin(__builtin_amdgcn_s_wait_asynccnt)
#  define USE_ASYNC 1
# endif
#endif
#ifndef USE_ASYNC
# define USE_ASYNC 0
#endif

typedef __attribute__((address_space(1))) v4i* v4i_g;
typedef __attribute__((address_space(3))) v4i* v4i_l;

// ---------------- batched bf16 WMMA GEMM (bf16 in-memory operands) ----------------
// C[m,n] = epilogue( scale * sum_k A[m,k] * Wop[n,k] )
//   normal:  Wop[n,k] = W[n*ldw + k]
//   transW:  Wop[n,k] = W[k*ldw + n]     (P @ V)
// FLAGS: 1=+bias 2=GELU 4=+Res 8=causal(n>m) 16=transW 32=triA 64=bf16 output
#define BM 128
#define BN 128
#define BK 32
#define LDP (BK + 8)   // halves; keeps 16B alignment for b128 LDS ops

template<int FLAGS>
__global__ __launch_bounds__(256) void gemm_bf16_wmma(
    const unsigned short* __restrict__ A, const unsigned short* __restrict__ W,
    const float* __restrict__ bias, const float* __restrict__ Res,
    void* __restrict__ Cout,
    int M, int N, int K, int lda, int ldw, int ldc,
    long long sA, long long sW, long long sC,
    float scale)
{
    constexpr bool HAS_BIAS = (FLAGS & 1) != 0;
    constexpr bool HAS_GELU = (FLAGS & 2) != 0;
    constexpr bool HAS_RES  = (FLAGS & 4) != 0;
    constexpr bool CAUSAL   = (FLAGS & 8) != 0;
    constexpr bool TRW      = (FLAGS & 16) != 0;
    constexpr bool TRIA     = (FLAGS & 32) != 0;
    constexpr bool OUT16    = (FLAGS & 64) != 0;

    const int n0 = blockIdx.x * BN;
    const int m0 = blockIdx.y * BM;
    const int bz = blockIdx.z;
    A += bz * sA;
    W += bz * sW;
    float*          Cf  = (float*)Cout + bz * sC;
    unsigned short* C16 = (unsigned short*)Cout + bz * sC;
    const float*    Rz  = HAS_RES ? Res + bz * sC : nullptr;

    // fully-masked causal tile: write -inf-ish, no math (block-uniform)
    if (CAUSAL && n0 > m0 + (BM - 1)) {
        for (int i = threadIdx.x; i < BM * BN; i += 256) {
            int r = i / BN, c = i % BN;
            Cf[(long long)(m0 + r) * ldc + n0 + c] = -3.0e38f;
        }
        return;
    }

    __shared__ unsigned short As[2][BM][LDP];
    __shared__ unsigned short Ws[2][BN][LDP];

    const int tid  = threadIdx.x;
    const int lane = tid & 31;
    const int wave = tid >> 5;           // 8 waves
    const int wm   = wave >> 2;          // 0..1 -> 64-row half
    const int wn   = wave & 3;           // 0..3 -> 32-col quarter
    const int half = lane >> 4;
    const int id16 = lane & 15;

    v8f acc[4][2];
    #pragma unroll
    for (int s = 0; s < 4; ++s)
        #pragma unroll
        for (int t = 0; t < 2; ++t) acc[s][t] = 0.f;

    int kend = K;
    if (TRIA) { int lim = m0 + BM; kend = K < lim ? K : lim; }

    // copy one 128x32-half tile (4096 halves = 512 x 16B chunks, 2 chunks/thread)
    auto copyA = [&](int buf, int k0) {
        #pragma unroll
        for (int j = 0; j < 2; ++j) {
            int idx = tid + j * 256;          // 0..511
            int r = idx >> 2, c8 = (idx & 3) * 8;
            const unsigned short* g = &A[(long long)(m0 + r) * lda + (k0 + c8)];
#if USE_ASYNC
            __builtin_amdgcn_global_load_async_to_lds_b128(
                (v4i_g)(void*)g, (v4i_l)(void*)&As[buf][r][c8], 0, 0);
#else
            *(uint4*)&As[buf][r][c8] = *(const uint4*)g;
#endif
        }
    };
    auto copyW = [&](int buf, int k0) {
        if (TRW) {
            // gather 8 halves along n, scatter to 8 LDS rows (P@V only; N==Dd==64 < BN)
            #pragma unroll
            for (int j = 0; j < 2; ++j) {
                int idx = tid + j * 256;
                int c = idx >> 4, n8 = (idx & 15) * 8;
                unsigned short h[8];
                if (n0 + n8 + 7 < N) {
                    uint4 d = *(const uint4*)&W[(long long)(k0 + c) * ldw + (n0 + n8)];
                    *(uint4*)h = d;
                } else {
                    #pragma unroll
                    for (int i = 0; i < 8; ++i) h[i] = 0;
                }
                #pragma unroll
                for (int i = 0; i < 8; ++i) Ws[buf][n8 + i][c] = h[i];
            }
        } else {
            #pragma unroll
            for (int j = 0; j < 2; ++j) {
                int idx = tid + j * 256;
                int r = idx >> 2, c8 = (idx & 3) * 8;
                const unsigned short* g = &W[(long long)(n0 + r) * ldw + (k0 + c8)];
#if USE_ASYNC
                __builtin_amdgcn_global_load_async_to_lds_b128(
                    (v4i_g)(void*)g, (v4i_l)(void*)&Ws[buf][r][c8], 0, 0);
#else
                *(uint4*)&Ws[buf][r][c8] = *(const uint4*)g;
#endif
            }
        }
    };

    copyA(0, 0);
    copyW(0, 0);
#if USE_ASYNC
    __builtin_amdgcn_s_wait_asynccnt(0);
#endif
    __syncthreads();

    int buf = 0;
    for (int k0 = 0; k0 < kend; k0 += BK) {
        const bool more = (k0 + BK) < kend;
        if (more) { copyA(buf ^ 1, k0 + BK); copyW(buf ^ 1, k0 + BK); }

        // fragments per ISA 7.12.2 16-bit layouts
        BF16x16 af[4], bw[2];
        #pragma unroll
        for (int s = 0; s < 4; ++s) {
            #pragma unroll
            for (int e = 0; e < 16; ++e) {
                int kk = (e < 8) ? (half * 8 + e) : (16 + half * 8 + (e - 8));
                af[s].u[e] = As[buf][wm * 64 + s * 16 + id16][kk];
            }
        }
        #pragma unroll
        for (int t = 0; t < 2; ++t) {
            #pragma unroll
            for (int e = 0; e < 16; ++e)
                bw[t].u[e] = Ws[buf][wn * 32 + t * 16 + id16][(e < 8) ? (half * 8 + e) : (16 + half * 8 + (e - 8))];
        }
        #pragma unroll
        for (int s = 0; s < 4; ++s)
            #pragma unroll
            for (int t = 0; t < 2; ++t)
                acc[s][t] = __builtin_amdgcn_wmma_f32_16x16x32_bf16(
                    false, af[s].v, false, bw[t].v, (short)0, acc[s][t], false, false);

        if (more) {
#if USE_ASYNC
            __builtin_amdgcn_s_wait_asynccnt(0);
#endif
            __syncthreads();
        }
        buf ^= 1;
    }

    // epilogue: VGPR r = row (strip + half*8 + r), col = id16
    #pragma unroll
    for (int s = 0; s < 4; ++s) {
        #pragma unroll
        for (int r = 0; r < 8; ++r) {
            int m = m0 + wm * 64 + s * 16 + half * 8 + r;
            #pragma unroll
            for (int t = 0; t < 2; ++t) {
                int n = n0 + wn * 32 + t * 16 + id16;
                if (TRW && n >= N) continue;   // only P@V has N < BN
                float v = acc[s][t][r] * scale;
                if (HAS_BIAS) v += bias[n];
                if (HAS_GELU) v = 0.5f * v * (1.0f + erff(v * 0.70710678118654752f));
                if (HAS_RES)  v += Rz[(long long)m * ldc + n];
                if (CAUSAL && n > m) v = -3.0e38f;
                if (OUT16) C16[(long long)m * ldc + n] = f2bfh(v);
                else       Cf [(long long)m * ldc + n] = v;
            }
        }
    }
}

// host-side typed launcher
template<int FLAGS>
static inline void gemm_t(hipStream_t stream,
                          const unsigned short* A, const unsigned short* W,
                          const float* bias, const float* Res, void* Cm,
                          int M, int N, int K, int lda, int ldw, int ldc,
                          long long sA, long long sW, long long sC, int batch,
                          float scale)
{
    dim3 grid((N + BN - 1) / BN, M / BM, batch);
    gemm_bf16_wmma<FLAGS><<<grid, dim3(256), 0, stream>>>(
        A, W, bias, Res, Cm, M, N, K, lda, ldw, ldc, sA, sW, sC, scale);
}

// ---------------- f32 -> bf16 bulk convert (weights etc.) ----------------
__global__ __launch_bounds__(256) void cvt_bf16_k(
    const float* __restrict__ src, unsigned short* __restrict__ dst, long long n)
{
    long long i = ((long long)blockIdx.x * 256 + threadIdx.x) * 8;
    if (i + 8 <= n) {
        float4 a = *(const float4*)(src + i);
        float4 b = *(const float4*)(src + i + 4);
        uint4 p;
        p.x = cvt2bf(a.x, a.y); p.y = cvt2bf(a.z, a.w);
        p.z = cvt2bf(b.x, b.y); p.w = cvt2bf(b.z, b.w);
        *(uint4*)(dst + i) = p;
    } else {
        for (long long j = i; j < n; ++j) dst[j] = f2bfh(src[j]);
    }
}

// ---------------- embedding (f32 out) ----------------
__global__ void embed_k(const int* __restrict__ idx,
                        const float* __restrict__ tok, const float* __restrict__ pos,
                        float* __restrict__ out)
{
    long long i = (long long)blockIdx.x * 256 + threadIdx.x;
    if (i >= (long long)MT * Cc) return;
    int c = (int)(i % Cc);
    int t = (int)((i / Cc) % Tt);
    int b = (int)(i / ((long long)Cc * Tt));
    int tk = idx[b * Tt + t];
    out[i] = tok[(long long)tk * Cc + c] + pos[(long long)t * Cc + c];
}

// ---------------- layernorm: f32 in, bf16 out; one wave per row ----------------
__global__ __launch_bounds__(256) void layernorm_k(
    const float* __restrict__ x, const float* __restrict__ g,
    const float* __restrict__ b, unsigned short* __restrict__ out, int rows, int n)
{
    int wave = threadIdx.x >> 5, lane = threadIdx.x & 31;
    int row = blockIdx.x * 8 + wave;
    if (row >= rows) return;
    const float* xr = x + (long long)row * n;
    float s = 0.f;
    for (int i = lane; i < n; i += 32) s += xr[i];
    float mean = wred_sum(s) / n;
    float v = 0.f;
    for (int i = lane; i < n; i += 32) { float d = xr[i] - mean; v += d * d; }
    float rstd = rsqrtf(wred_sum(v) / n + 1e-5f);
    unsigned short* orow = out + (long long)row * n;
    for (int i = lane; i < n; i += 32)
        orow[i] = f2bfh((xr[i] - mean) * rstd * g[i] + b[i]);
}

// ---------------- softmax: f32 scores in, bf16 probs out ----------------
__global__ __launch_bounds__(256) void softmax_k(
    float* __restrict__ a, unsigned short* __restrict__ out, int rows, int n)
{
    int wave = threadIdx.x >> 5, lane = threadIdx.x & 31;
    int row = blockIdx.x * 8 + wave;
    if (row >= rows) return;
    float* ar = a + (long long)row * n;
    unsigned short* orow = out + (long long)row * n;
    float mx = -3.4e38f;
    for (int i = lane; i < n; i += 32) mx = fmaxf(mx, ar[i]);
    mx = wred_max(mx);
    float s = 0.f;
    for (int i = lane; i < n; i += 32) { float e = __expf(ar[i] - mx); ar[i] = e; s += e; }
    s = wred_sum(s);
    float inv = 1.0f / s;
    for (int i = lane; i < n; i += 32) orow[i] = f2bfh(ar[i] * inv);
}

// ---------------- cross-entropy from logits ----------------
__global__ void zero_loss_k(float* loss) { if (threadIdx.x == 0) *loss = 0.f; }

__global__ __launch_bounds__(256) void rowloss_k(
    const float* __restrict__ logits, const int* __restrict__ tgt,
    float* __restrict__ loss, int rows, int n)
{
    int wave = threadIdx.x >> 5, lane = threadIdx.x & 31;
    int row = blockIdx.x * 8 + wave;
    if (row >= rows) return;
    const float* lr = logits + (long long)row * n;
    float mx = -3.4e38f;
    for (int i = lane; i < n; i += 32) mx = fmaxf(mx, lr[i]);
    mx = wred_max(mx);
    float s = 0.f;
    for (int i = lane; i < n; i += 32) s += __expf(lr[i] - mx);
    s = wred_sum(s);
    if (lane == 0) {
        float lt = lr[tgt[row]];
        atomicAdd(loss, (logf(s) + mx - lt) / (float)rows);
    }
}

// ---------------- host-side orchestration ----------------
extern "C" void kernel_launch(void* const* d_in, const int* in_sizes, int n_in,
                              void* d_out, int out_size, void* d_ws, size_t ws_size,
                              hipStream_t stream)
{
    (void)in_sizes; (void)n_in; (void)out_size; (void)ws_size;

    const int* idx     = (const int*)d_in[0];
    const int* targets = (const int*)d_in[1];
    // JAX pytree flatten order (dict keys sorted):
    // blocks (8 x {b1,b2,ln1_b,ln1_g,ln2_b,ln2_g,w1,w2,wk,wo,wq,wv}), head, lnf_b, lnf_g, pos_emb, tok_emb
    auto blkp = [&](int l, int j) { return (const float*)d_in[2 + l * 12 + j]; };
    const float* headw   = (const float*)d_in[2 + 96 + 0];
    const float* lnf_b   = (const float*)d_in[2 + 96 + 1];
    const float* lnf_g   = (const float*)d_in[2 + 96 + 2];
    const float* pos_emb = (const float*)d_in[2 + 96 + 3];
    const float* tok_emb = (const float*)d_in[2 + 96 + 4];

    // workspace (byte cursor, 256B aligned slices)
    char* base = (char*)d_ws;
    size_t off = 0;
    auto alloc = [&](size_t bytes) {
        void* p = base + off;
        off += (bytes + 255) & ~(size_t)255;
        return p;
    };
    float* x    = (float*)alloc((size_t)MT * Cc * 4);
    float* x2   = (float*)alloc((size_t)MT * Cc * 4);
    float* attf = (float*)alloc((size_t)Hh * Tt * Tt * 4);
    unsigned short* hb   = (unsigned short*)alloc((size_t)MT * Cc * 2);
    unsigned short* q    = (unsigned short*)alloc((size_t)MT * Cc * 2);
    unsigned short* k    = (unsigned short*)alloc((size_t)MT * Cc * 2);
    unsigned short* v    = (unsigned short*)alloc((size_t)MT * Cc * 2);
    unsigned short* y    = (unsigned short*)alloc((size_t)MT * Cc * 2);
    unsigned short* h1   = (unsigned short*)alloc((size_t)MT * FOURC * 2);
    unsigned short* attb = (unsigned short*)alloc((size_t)Hh * Tt * Tt * 2);
    unsigned short* wqb  = (unsigned short*)alloc((size_t)Cc * Cc * 2);
    unsigned short* wkb  = (unsigned short*)alloc((size_t)Cc * Cc * 2);
    unsigned short* wvb  = (unsigned short*)alloc((size_t)Cc * Cc * 2);
    unsigned short* wob  = (unsigned short*)alloc((size_t)Cc * Cc * 2);
    unsigned short* w1b  = (unsigned short*)alloc((size_t)FOURC * Cc * 2);
    unsigned short* w2b  = (unsigned short*)alloc((size_t)FOURC * Cc * 2);
    unsigned short* hdb  = (unsigned short*)alloc((size_t)Vv * Cc * 2);

    dim3 blk256(256);
    auto cvt = [&](const float* s, unsigned short* d, long long n) {
        cvt_bf16_k<<<(int)((n / 8 + 255) / 256), blk256, 0, stream>>>(s, d, n);
    };

    embed_k<<<(int)(((long long)MT * Cc + 255) / 256), blk256, 0, stream>>>(idx, tok_emb, pos_emb, x);

    const float att_scale = 0.125f;   // 1/sqrt(64)

    for (int l = 0; l < Ll; ++l) {
        const float* b1   = blkp(l, 0);
        const float* b2   = blkp(l, 1);
        const float* ln1b = blkp(l, 2);
        const float* ln1g = blkp(l, 3);
        const float* ln2b = blkp(l, 4);
        const float* ln2g = blkp(l, 5);

        // weights -> bf16 (stream-ordered, reused buffers)
        cvt(blkp(l, 10), wqb, (long long)Cc * Cc);    // wq
        cvt(blkp(l, 8),  wkb, (long long)Cc * Cc);    // wk
        cvt(blkp(l, 11), wvb, (long long)Cc * Cc);    // wv
        cvt(blkp(l, 9),  wob, (long long)Cc * Cc);    // wo
        cvt(blkp(l, 6),  w1b, (long long)FOURC * Cc); // w1
        cvt(blkp(l, 7),  w2b, (long long)FOURC * Cc); // w2

        layernorm_k<<<MT / 8, blk256, 0, stream>>>(x, ln1g, ln1b, hb, MT, Cc);
        // q,k,v (bf16 out)
        gemm_t<64>(stream, hb, wqb, nullptr, nullptr, q, MT, Cc, Cc, Cc, Cc, Cc, 0, 0, 0, 1, 1.f);
        gemm_t<64>(stream, hb, wkb, nullptr, nullptr, k, MT, Cc, Cc, Cc, Cc, Cc, 0, 0, 0, 1, 1.f);
        gemm_t<64>(stream, hb, wvb, nullptr, nullptr, v, MT, Cc, Cc, Cc, Cc, Cc, 0, 0, 0, 1, 1.f);

        for (int b = 0; b < Bb; ++b) {
            const unsigned short* qb = q + (size_t)b * Tt * Cc;
            const unsigned short* kb = k + (size_t)b * Tt * Cc;
            const unsigned short* vb = v + (size_t)b * Tt * Cc;
            unsigned short* yb = y + (size_t)b * Tt * Cc;
            // scores (f32 out for softmax, causal)
            gemm_t<8>(stream, qb, kb, nullptr, nullptr, attf, Tt, Tt, Dd, Cc, Cc, Tt,
                      Dd, Dd, (long long)Tt * Tt, Hh, att_scale);
            softmax_k<<<(Hh * Tt) / 8, blk256, 0, stream>>>(attf, attb, Hh * Tt, Tt);
            // y = P @ V (transW + triA + bf16 out)
            gemm_t<16 | 32 | 64>(stream, attb, vb, nullptr, nullptr, yb, Tt, Dd, Tt, Tt, Cc, Cc,
                                 (long long)Tt * Tt, Dd, Dd, Hh, 1.f);
        }

        // x2 = x + y @ wo.T (residual, f32 out)
        gemm_t<4>(stream, y, wob, nullptr, x, x2, MT, Cc, Cc, Cc, Cc, Cc, 0, 0, 0, 1, 1.f);
        layernorm_k<<<MT / 8, blk256, 0, stream>>>(x2, ln2g, ln2b, hb, MT, Cc);
        // h1 = gelu(h @ w1.T + b1) (bf16 out)
        gemm_t<1 | 2 | 64>(stream, hb, w1b, b1, nullptr, h1, MT, FOURC, Cc, Cc, Cc, FOURC, 0, 0, 0, 1, 1.f);
        // x = x2 + h1 @ w2.T + b2 (f32 out)
        gemm_t<1 | 4>(stream, h1, w2b, b2, x2, x, MT, Cc, FOURC, FOURC, FOURC, Cc, 0, 0, 0, 1, 1.f);
    }

    layernorm_k<<<MT / 8, blk256, 0, stream>>>(x, lnf_g, lnf_b, hb, MT, Cc);
    cvt(headw, hdb, (long long)Vv * Cc);
    float* logits = (float*)d_out;
    gemm_t<0>(stream, hb, hdb, nullptr, nullptr, logits, MT, Vv, Cc, Cc, Cc, Vv, 0, 0, 0, 1, 1.f);

    float* loss = logits + (size_t)MT * Vv;
    zero_loss_k<<<1, 32, 0, stream>>>(loss);
    rowloss_k<<<MT / 8, blk256, 0, stream>>>(logits, targets, loss, MT, Vv);
}
